// RasterizePointsXYsBlending_34419867910824
// MI455X (gfx1250) — matrix operand order repaired and checked by reading, not compiled, not run on hardware
//
#include <hip/hip_runtime.h>
#include <stdint.h>

typedef _Float16 half_t;
typedef __attribute__((ext_vector_type(16))) _Float16 v16h;
typedef __attribute__((ext_vector_type(8)))  float    v8f;

#define BATCH 2
#define NPTS  65536
#define NCH   64
#define HH    352
#define WW    1216
#define KD    8
#define NPIX  (BATCH * HH * WW)          // 856064, divisible by 16
#define NTILE (NPIX / 16)                // 53504

// workspace layout (bytes)
#define CNT_OFF 0u
#define IDX_OFF 3424256u                                   // NPIX*4
#define WB_OFF  (3424256u + 27394048u)                     // + NPIX*8*4
#define F16_OFF (3424256u + 27394048u + 27394048u)         // + NPIX*8*4
#define ZP_OFF  (3424256u + 27394048u + 27394048u + 16777216u)  // + B*P*64*2 ; 128B zero page

// ---------------------------------------------------------------- pass 0a: zero counters + zero page
__global__ void zero_counts_kernel(unsigned* __restrict__ cnt, unsigned* __restrict__ zp) {
    int t = blockIdx.x * blockDim.x + threadIdx.x;
    if (t < NPIX) cnt[t] = 0u;
    if (t < 32) zp[t] = 0u;   // 128-byte zero page for empty-slot async copies
}

// ---------------------------------------------------------------- pass 0b: src [B,C,P] f32 -> feats [B*P, C] f16
__global__ __launch_bounds__(256) void transpose_feats_kernel(const float* __restrict__ src,
                                                              half_t* __restrict__ dst) {
    __shared__ float tile[64][65];
    const int b  = blockIdx.y;
    const int p0 = blockIdx.x * 64;
    const int t  = threadIdx.x;
    const int px = t & 63;
    const int r0 = t >> 6;                 // 0..3
    #pragma unroll
    for (int c = r0; c < 64; c += 4)
        tile[c][px] = src[((size_t)b * NCH + c) * NPTS + p0 + px];
    __syncthreads();
    #pragma unroll
    for (int pp = r0; pp < 64; pp += 4)
        dst[((size_t)(b * NPTS + p0 + pp)) * NCH + px] = (half_t)tile[px][pp];
}

// ---------------------------------------------------------------- pass 1: rasterize (scatter-append)
__global__ __launch_bounds__(256) void rasterize_kernel(const float* __restrict__ pts,
                                                        unsigned* __restrict__ cnt,
                                                        int* __restrict__ idxb,
                                                        float* __restrict__ wb) {
    const int t = blockIdx.x * blockDim.x + threadIdx.x;
    if (t >= BATCH * NPTS) return;
    const int b = t / NPTS;
    const float x = pts[(size_t)t * 3 + 0];
    const float y = pts[(size_t)t * 3 + 1];
    const float z = pts[(size_t)t * 3 + 2];
    if (z <= 0.0f) return;
    const float r   = 3.0f / 1216.0f;      // RADIUS_PX/max(H,W)*2
    const float r2  = r * r;
    const float ir2 = 1.0f / r2;
    const int i0 = (int)floorf((y + 1.0f) * 0.5f * (float)HH);
    const int j0 = (int)floorf((x + 1.0f) * 0.5f * (float)WW);
    #pragma unroll
    for (int dyy = -1; dyy <= 1; ++dyy) {
        const int ii = i0 + dyy;
        if (ii < 0 || ii >= HH) continue;
        const float yc = ((float)ii + 0.5f) * (2.0f / (float)HH) - 1.0f;
        const float dy2 = (y - yc) * (y - yc);
        #pragma unroll
        for (int dxx = -2; dxx <= 2; ++dxx) {
            const int jj = j0 + dxx;
            if (jj < 0 || jj >= WW) continue;
            const float xc = ((float)jj + 0.5f) * (2.0f / (float)WW) - 1.0f;
            const float d2 = (x - xc) * (x - xc) + dy2;
            if (d2 < r2) {
                const int pix = (b * HH + ii) * WW + jj;
                const unsigned slot = atomicAdd(&cnt[pix], 1u);
                if (slot < KD) {
                    float dist = d2 * ir2;                    // rad_pow = 2
                    dist = fminf(fmaxf(dist, 0.001f), 1.0f);
                    const float alpha = 1.0f - sqrtf(dist);   // tau = 1
                    idxb[pix * KD + slot] = t;
                    wb[pix * KD + slot]   = alpha;
                }
            }
        }
    }
}

// ---------------------------------------------------------------- pass 2: WMMA composite
// One wave per 16-pixel tile. D[ch,pix] = A(16ch x 32slot, f16 feats) * B(32slot x 16pix, block-diag weights)
// Feature rows staged with GLOBAL_LOAD_ASYNC_TO_LDS_B128 (empty slots -> zero page).
__global__ __launch_bounds__(32) void composite_kernel(const unsigned* __restrict__ cnt,
                                                       const int* __restrict__ idxb,
                                                       const float* __restrict__ wb,
                                                       const half_t* __restrict__ feats,
                                                       const half_t* __restrict__ zp,
                                                       float* __restrict__ out) {
    // LDS: row r = p*8+k (16 pixels x 8 slots), 64 f16 channels per row = 16KB
    __shared__ __align__(16) half_t srows[128 * 64];
    const int lane = threadIdx.x;
    const int pix0 = blockIdx.x * 16;
    const int b    = pix0 / (HH * WW);
    const int rem  = pix0 - b * (HH * WW);
    const int irow = rem / WW;
    const int jcol = rem - irow * WW;

    // ---- async-stage gathered feature rows into LDS (2 lanes per pixel, 4 slots each)
    {
        const int p  = lane >> 1;
        const int kb = (lane & 1) * 4;
        const int pg = pix0 + p;
        const unsigned c = cnt[pg];
        const int n = (c < (unsigned)KD) ? (int)c : KD;
        const int4 iv = *(const int4*)&idxb[pg * KD + kb];
        const half_t* bases[4];
        bases[0] = (kb + 0 < n) ? (feats + (size_t)iv.x * NCH) : zp;
        bases[1] = (kb + 1 < n) ? (feats + (size_t)iv.y * NCH) : zp;
        bases[2] = (kb + 2 < n) ? (feats + (size_t)iv.z * NCH) : zp;
        bases[3] = (kb + 3 < n) ? (feats + (size_t)iv.w * NCH) : zp;
        const int rbase = p * KD + kb;
        #pragma unroll
        for (int rr = 0; rr < 4; ++rr) {
            const unsigned ldsa = (unsigned)(uintptr_t)&srows[(rbase + rr) * NCH];
            const char* g = (const char*)bases[rr];
            // INST_OFFSET applies to both global source and LDS destination (ISA 10.x async pseudocode)
            asm volatile("global_load_async_to_lds_b128 %0, %1, off"            :: "v"(ldsa), "v"(g) : "memory");
            asm volatile("global_load_async_to_lds_b128 %0, %1, off offset:16"  :: "v"(ldsa), "v"(g) : "memory");
            asm volatile("global_load_async_to_lds_b128 %0, %1, off offset:32"  :: "v"(ldsa), "v"(g) : "memory");
            asm volatile("global_load_async_to_lds_b128 %0, %1, off offset:48"  :: "v"(ldsa), "v"(g) : "memory");
            asm volatile("global_load_async_to_lds_b128 %0, %1, off offset:64"  :: "v"(ldsa), "v"(g) : "memory");
            asm volatile("global_load_async_to_lds_b128 %0, %1, off offset:80"  :: "v"(ldsa), "v"(g) : "memory");
            asm volatile("global_load_async_to_lds_b128 %0, %1, off offset:96"  :: "v"(ldsa), "v"(g) : "memory");
            asm volatile("global_load_async_to_lds_b128 %0, %1, off offset:112" :: "v"(ldsa), "v"(g) : "memory");
        }
    }
    asm volatile("s_wait_asynccnt 0" ::: "memory");
    __syncthreads();

    const int chlocal = lane & 15;
    const int hi = lane >> 4;
    // B-fragment holder lanes: column n<8 -> lane n; column n>=8 -> lane n+16
    const bool holder = (lane < 8) || (lane >= 24);
    const int  ncol   = lane & 15;
    const int  vhold  = ncol & 7;
    const int  pgn    = pix0 + ncol;
    const unsigned cN = cnt[pgn];
    const int cntn = (cN < (unsigned)KD) ? (int)cN : KD;

    // unconditional vectorized weight load, gated by selects (no divergent branches)
    const float4 wlo = *(const float4*)&wb[pgn * KD];
    const float4 whi = *(const float4*)&wb[pgn * KD + 4];
    float wk[8] = {wlo.x, wlo.y, wlo.z, wlo.w, whi.x, whi.y, whi.z, whi.w};
    #pragma unroll
    for (int k = 0; k < 8; ++k)
        wk[k] = (holder && k < cntn) ? wk[k] : 0.0f;

    v8f acc[4] = {};
    #pragma unroll
    for (int q = 0; q < 4; ++q) {
        // ---- block-diagonal B (32x16 f16): B[2n+s, n] = w[n, 2q+s]
        union { half_t h[2]; unsigned u; } pw;
        pw.h[0] = (half_t)wk[2 * q];
        pw.h[1] = (half_t)wk[2 * q + 1];
        union { v16h v; unsigned u32[8]; } bu;
        #pragma unroll
        for (int vv = 0; vv < 8; ++vv)
            bu.u32[vv] = (holder && vv == vhold) ? pw.u : 0u;

        // ---- A fragments (16ch x 32slot) per channel chunk, from LDS rows
        #pragma unroll
        for (int mc = 0; mc < 4; ++mc) {
            const int ch = mc * 16 + chlocal;
            union { v16h v; half_t h[16]; } au;
            #pragma unroll
            for (int tv = 0; tv < 8; ++tv) {
                // VGPR tv holds K pair: K = hi*16 + 2*tv -> pixel pe, slots 2q/2q+1
                const int pe = hi * 8 + tv;
                au.h[2 * tv]     = srows[(pe * KD + 2 * q) * NCH + ch];
                au.h[2 * tv + 1] = srows[(pe * KD + 2 * q + 1) * NCH + ch];
            }
            acc[mc] = __builtin_amdgcn_wmma_f32_16x16x32_f16(
                false, au.v, false, bu.v, (short)0, acc[mc], false, false);
        }
    }

    // ---- store D: lane holds column n=lane&15 (pixel), VGPR vv holds row m=vv+8*hi (channel in chunk)
    #pragma unroll
    for (int mc = 0; mc < 4; ++mc) {
        #pragma unroll
        for (int vv = 0; vv < 8; ++vv) {
            const int ch = mc * 16 + vv + 8 * hi;
            out[(((size_t)b * NCH + ch) * HH + irow) * WW + (jcol + ncol)] = acc[mc][vv];
        }
    }
}

// ---------------------------------------------------------------- launcher
extern "C" void kernel_launch(void* const* d_in, const int* in_sizes, int n_in,
                              void* d_out, int out_size, void* d_ws, size_t ws_size,
                              hipStream_t stream) {
    const float* pts = (const float*)d_in[0];   // [B,P,3] f32
    const float* src = (const float*)d_in[1];   // [B,C,P] f32
    float* out = (float*)d_out;                 // [B,C,H,W] f32

    uint8_t* ws = (uint8_t*)d_ws;
    unsigned* cnt  = (unsigned*)(ws + CNT_OFF);
    int*      idxb = (int*)(ws + IDX_OFF);
    float*    wb   = (float*)(ws + WB_OFF);
    half_t*   f16  = (half_t*)(ws + F16_OFF);
    unsigned* zp   = (unsigned*)(ws + ZP_OFF);

    zero_counts_kernel<<<(NPIX + 255) / 256, 256, 0, stream>>>(cnt, zp);
    transpose_feats_kernel<<<dim3(NPTS / 64, BATCH), 256, 0, stream>>>(src, f16);
    rasterize_kernel<<<(BATCH * NPTS + 255) / 256, 256, 0, stream>>>(pts, cnt, idxb, wb);
    composite_kernel<<<NTILE, 32, 0, stream>>>(cnt, idxb, wb, f16, (const half_t*)zp, out);
}
